// KGPathGNNDecoderModel_481036337933
// MI455X (gfx1250) — matrix operand order repaired and checked by reading, not compiled, not run on hardware
//
#include <hip/hip_runtime.h>
#include <hip/hip_bf16.h>
#include <math.h>

// ---------------- model constants ----------------
#define N_NODES  16384
#define NODE_DIM 256
#define NEDGE    262144
#define BATCH    16
#define LSEQQ    24
#define KCTX     128
#define RREL     2000
#define DMODEL   256
#define NHEAD    8
#define HDIM     32
#define N_GNN    3
#define N_DEC    6
#define ALPHA    0.2f
#define NTOK     (BATCH * LSEQQ)   // 384

typedef __attribute__((ext_vector_type(16))) _Float16 v16h;
typedef __attribute__((ext_vector_type(8)))  _Float16 v8h;
typedef __attribute__((ext_vector_type(8)))  float    v8f;

// =====================================================================
// Generic WMMA GEMM:  C[M,Nn] = A[M,Kd] @ W[Nn,Kd]^T (+bias)(+gelu)(+resid)
// Block = 256 threads (8 wave32s). Block tile 128(M) x 64(N); wave tile
// 32x32 (4 accumulators -> 4 x v_wmma_f32_16x16x32_f16 per K-step of 32).
// A/W staged in LDS as f16 via float4 loads + v8h stores; f32 accumulation.
// Requirements: M % 128 == 0, Kd % 32 == 0 (true for all call sites);
// Nn may be ragged (head Nn=2000): W row addresses are clamped, and the
// epilogue guards stores, so clamped rows only feed dead accumulator cols.
// =====================================================================
#define TILEM    128
#define TILEN    64
#define KSTEP    32
#define LDSS     40   // padded f16 leading dim (bytes stride 80 = 5*16, keeps 16B alignment)

__global__ __launch_bounds__(256)
void gemm_wmma_kernel(const float* __restrict__ A, const float* __restrict__ W,
                      const float* __restrict__ bias, const float* __restrict__ resid,
                      float* __restrict__ C, int M, int Nn, int Kd, int epi)
{
    __shared__ _Float16 As[TILEM * LDSS];
    __shared__ _Float16 Ws[TILEN * LDSS];

    const int tid  = threadIdx.x;
    const int lane = tid & 31;
    const int wv   = tid >> 5;
    const int wm   = wv & 3;     // 4 wave-rows (32 M each)
    const int wn   = wv >> 2;    // 2 wave-cols (32 N each)
    const int m0   = blockIdx.y * TILEM;
    const int n0   = blockIdx.x * TILEN;

    v8f acc00 = {}, acc01 = {}, acc10 = {}, acc11 = {};

    // staging coordinates (8 consecutive floats per thread per round)
    const int sr = tid >> 2;          // A: rows via 2 rounds; W: rows 0..63
    const int sc = (tid & 3) * 8;     // col start 0/8/16/24

    for (int k0 = 0; k0 < Kd; k0 += KSTEP) {
        // ---- stage A tile 128x32 (two rounds of 256 threads x 8 floats) ----
        #pragma unroll
        for (int t = 0; t < 2; ++t) {
            int r = sr + t * 64;
            const float* ap = A + (size_t)min(m0 + r, M - 1) * Kd + k0 + sc;
            float4 q0 = ((const float4*)ap)[0];
            float4 q1 = ((const float4*)ap)[1];
            v8h hv;
            hv[0] = (_Float16)q0.x; hv[1] = (_Float16)q0.y;
            hv[2] = (_Float16)q0.z; hv[3] = (_Float16)q0.w;
            hv[4] = (_Float16)q1.x; hv[5] = (_Float16)q1.y;
            hv[6] = (_Float16)q1.z; hv[7] = (_Float16)q1.w;
            *(v8h*)&As[r * LDSS + sc] = hv;
        }
        // ---- stage W tile 64x32 (one round) ----
        {
            const float* wp = W + (size_t)min(n0 + sr, Nn - 1) * Kd + k0 + sc;
            float4 q0 = ((const float4*)wp)[0];
            float4 q1 = ((const float4*)wp)[1];
            v8h hv;
            hv[0] = (_Float16)q0.x; hv[1] = (_Float16)q0.y;
            hv[2] = (_Float16)q0.z; hv[3] = (_Float16)q0.w;
            hv[4] = (_Float16)q1.x; hv[5] = (_Float16)q1.y;
            hv[6] = (_Float16)q1.z; hv[7] = (_Float16)q1.w;
            *(v8h*)&Ws[sr * LDSS + sc] = hv;
        }
        __syncthreads();

        // ---- gather WMMA fragments per ISA 16-bit layouts ----
        v16h a0, a1, b0, b1;
        const int am   = (wm << 5) + (lane & 15);
        const int aoff = (lane >> 4) << 3;          // lanes16-31 start at K=8
        #pragma unroll
        for (int j = 0; j < 16; ++j) {
            int kk = ((j >> 3) << 4) + aoff + (j & 7);
            a0[j] = As[(am)      * LDSS + kk];
            a1[j] = As[(am + 16) * LDSS + kk];
        }
        const int bn   = (wn << 5) + (lane & 15);
        const int koff = (lane >> 4) << 3;          // lanes16-31 start at K=8
        #pragma unroll
        for (int j = 0; j < 16; ++j) {
            int kk = ((j >> 3) << 4) + koff + (j & 7);
            b0[j] = Ws[(bn)      * LDSS + kk];
            b1[j] = Ws[(bn + 16) * LDSS + kk];
        }

        acc00 = __builtin_amdgcn_wmma_f32_16x16x32_f16(false, a0, false, b0, (short)0, acc00, false, false);
        acc01 = __builtin_amdgcn_wmma_f32_16x16x32_f16(false, a0, false, b1, (short)0, acc01, false, false);
        acc10 = __builtin_amdgcn_wmma_f32_16x16x32_f16(false, a1, false, b0, (short)0, acc10, false, false);
        acc11 = __builtin_amdgcn_wmma_f32_16x16x32_f16(false, a1, false, b1, (short)0, acc11, false, false);
        __syncthreads();
    }

    // ---- epilogue: C layout VGPR i -> M = base + (lane>=16 ? 8:0)+i, N = lane&15 ----
    const int rb = m0 + (wm << 5) + ((lane >> 4) << 3);
    const int cb = n0 + (wn << 5) + (lane & 15);
    #pragma unroll
    for (int si = 0; si < 2; ++si) {
        #pragma unroll
        for (int sj = 0; sj < 2; ++sj) {
            v8f acc = si ? (sj ? acc11 : acc10) : (sj ? acc01 : acc00);
            int col = cb + sj * 16;
            if (col >= Nn) continue;
            float bv = bias ? bias[col] : 0.0f;
            #pragma unroll
            for (int i = 0; i < 8; ++i) {
                int row = rb + si * 16 + i;
                float v = acc[i] + bv;
                if (epi == 1) {  // tanh-approx GELU (jax.nn.gelu default)
                    float x3 = v * v * v;
                    v = 0.5f * v * (1.0f + tanhf(0.7978845608028654f * (v + 0.044715f * x3)));
                }
                if (resid) v += resid[(size_t)row * Nn + col];
                C[(size_t)row * Nn + col] = v;
            }
        }
    }
}

// =====================================================================
// GAT helper kernels
// =====================================================================
__device__ inline unsigned fenc(float f) {
    unsigned u = __float_as_uint(f);
    return (u & 0x80000000u) ? ~u : (u | 0x80000000u);   // order-preserving
}
__device__ inline float fdec(unsigned e) {
    unsigned u = (e & 0x80000000u) ? (e & 0x7fffffffu) : ~e;
    return __uint_as_float(u);
}

// s1[n,h] = sum_d h[n,h,d]*a[d]; s2 with a[HD:]  (wave per (node,head))
__global__ __launch_bounds__(256)
void gat_score_kernel(const float* __restrict__ hfeat, const float* __restrict__ a,
                      float* __restrict__ s1, float* __restrict__ s2)
{
    int n = blockIdx.x;
    int h = threadIdx.x >> 5;
    int d = threadIdx.x & 31;
    float hv = hfeat[n * DMODEL + h * HDIM + d];
    float v1 = hv * a[d];
    float v2 = hv * a[HDIM + d];
    #pragma unroll
    for (int o = 16; o > 0; o >>= 1) {
        v1 += __shfl_xor(v1, o, 32);
        v2 += __shfl_xor(v2, o, 32);
    }
    if (d == 0) { s1[n * NHEAD + h] = v1; s2[n * NHEAD + h] = v2; }
}

__global__ void gat_init_kernel(unsigned* menc, float* sumex, unsigned* deg, float* gout)
{
    int i = blockIdx.x * 256 + threadIdx.x;
    if (i < N_NODES * NHEAD) { menc[i] = fenc(-3.4028235e38f); sumex[i] = 0.0f; }
    if (i < N_NODES)          deg[i] = 0u;
    if (i < N_NODES * DMODEL) gout[i] = 0.0f;
}

// pass 1: leaky score per (edge,head) + encoded atomic segment-max + degree
__global__ void edge_score_kernel(const int* __restrict__ src, const int* __restrict__ tgt,
                                  const float* __restrict__ s1, const float* __restrict__ s2,
                                  float* __restrict__ score, unsigned* __restrict__ menc,
                                  unsigned* __restrict__ deg)
{
    int i = blockIdx.x * 256 + threadIdx.x;
    if (i >= NEDGE * NHEAD) return;
    int e = i >> 3, h = i & 7;
    int s = src[e], t = tgt[e];
    float v = s1[s * NHEAD + h] + s2[t * NHEAD + h];
    v = v > 0.0f ? v : ALPHA * v;
    score[i] = v;
    atomicMax(&menc[s * NHEAD + h], fenc(v));
    if (h == 0) atomicAdd(&deg[s], 1u);
}

// decode encoded max and clamp to 0 (implicit-zero entries), store as float in-place
__global__ void gat_mclamp_kernel(unsigned* menc)
{
    int i = blockIdx.x * 256 + threadIdx.x;
    if (i >= N_NODES * NHEAD) return;
    float m = fmaxf(fdec(menc[i]), 0.0f);
    ((float*)menc)[i] = m;
}

// pass 2: ex = exp(e - m[src]); accumulate segment sum
__global__ void edge_exp_kernel(const int* __restrict__ src, float* __restrict__ score,
                                const float* __restrict__ m, float* __restrict__ sumex)
{
    int i = blockIdx.x * 256 + threadIdx.x;
    if (i >= NEDGE * NHEAD) return;
    int e = i >> 3, h = i & 7;
    int s = src[e];
    float ex = expf(score[i] - m[s * NHEAD + h]);
    score[i] = ex;
    atomicAdd(&sumex[s * NHEAD + h], ex);
}

// denom[n,h] = (N - deg)*exp(-m) + sumex
__global__ void gat_denom_kernel(const float* __restrict__ m, const float* __restrict__ sumex,
                                 const unsigned* __restrict__ deg, float* __restrict__ denom)
{
    int i = blockIdx.x * 256 + threadIdx.x;
    if (i >= N_NODES * NHEAD) return;
    int n = i >> 3;
    denom[i] = ((float)N_NODES - (float)deg[n]) * expf(-m[i]) + sumex[i];
}

// aggregate: wave per edge, lane per feature within head; 8 heads per lane
__global__ __launch_bounds__(256)
void gat_aggregate_kernel(const int* __restrict__ src, const int* __restrict__ tgt,
                          const float* __restrict__ score, const float* __restrict__ denom,
                          const float* __restrict__ hfeat, float* __restrict__ gout)
{
    int wv = threadIdx.x >> 5, lane = threadIdx.x & 31;
    int e = blockIdx.x * 8 + wv;
    if (e >= NEDGE) return;
    int s = src[e], t = tgt[e];
    #pragma unroll
    for (int h = 0; h < NHEAD; ++h) {
        float coef = score[e * NHEAD + h] / denom[s * NHEAD + h];
        atomicAdd(&gout[s * DMODEL + h * HDIM + lane],
                  coef * hfeat[t * DMODEL + h * HDIM + lane]);
    }
}

// x = LN(x + (deg>0 ? gout : hfeat)); block of 256 = one node
__global__ __launch_bounds__(256)
void gat_post_kernel(float* __restrict__ x, const float* __restrict__ gout,
                     const float* __restrict__ hfeat, const unsigned* __restrict__ deg,
                     const float* __restrict__ g, const float* __restrict__ b)
{
    __shared__ float r1[256];
    __shared__ float r2[256];
    int n = blockIdx.x, d = threadIdx.x;
    float o = (deg[n] > 0u) ? gout[n * DMODEL + d] : hfeat[n * DMODEL + d];
    float v = x[n * DMODEL + d] + o;
    r1[d] = v; r2[d] = v * v;
    __syncthreads();
    for (int s2 = 128; s2 > 0; s2 >>= 1) {
        if (d < s2) { r1[d] += r1[d + s2]; r2[d] += r2[d + s2]; }
        __syncthreads();
    }
    float mean = r1[0] * (1.0f / DMODEL);
    float var  = r2[0] * (1.0f / DMODEL) - mean * mean;
    x[n * DMODEL + d] = (v - mean) * rsqrtf(var + 1e-5f) * g[d] + b[d];
}

// =====================================================================
// Decoder helper kernels
// =====================================================================
// Row LayerNorm, wave per row (D=256, 8 elements/lane)
__global__ __launch_bounds__(256)
void ln_rows_kernel(const float* __restrict__ in, float* __restrict__ out,
                    const float* __restrict__ g, const float* __restrict__ b, int rows)
{
    int wv = threadIdx.x >> 5, lane = threadIdx.x & 31;
    int row = blockIdx.x * 8 + wv;
    if (row >= rows) return;
    float vals[8]; float s = 0.0f, s2 = 0.0f;
    #pragma unroll
    for (int i = 0; i < 8; ++i) {
        float v = in[row * DMODEL + i * 32 + lane];
        vals[i] = v; s += v; s2 += v * v;
    }
    #pragma unroll
    for (int o = 16; o > 0; o >>= 1) { s += __shfl_xor(s, o, 32); s2 += __shfl_xor(s2, o, 32); }
    float mean = s * (1.0f / DMODEL);
    float var  = s2 * (1.0f / DMODEL) - mean * mean;
    float rs = rsqrtf(var + 1e-5f);
    #pragma unroll
    for (int i = 0; i < 8; ++i)
        out[row * DMODEL + i * 32 + lane] = (vals[i] - mean) * rs * g[i * 32 + lane] + b[i * 32 + lane];
}

// causal self-attention, block per (batch, head); qkv ld = 768
__global__ __launch_bounds__(256)
void self_attn_kernel(const float* __restrict__ qkv, float* __restrict__ out)
{
    __shared__ float qs[LSEQQ][HDIM], ks[LSEQQ][HDIM], vs[LSEQQ][HDIM];
    __shared__ float sc[LSEQQ][LSEQQ];
    int b = blockIdx.x >> 3, h = blockIdx.x & 7;
    int tid = threadIdx.x;
    for (int idx = tid; idx < LSEQQ * HDIM; idx += 256) {
        int t = idx >> 5, d = idx & 31;
        const float* row = qkv + (size_t)(b * LSEQQ + t) * (3 * DMODEL);
        qs[t][d] = row[h * HDIM + d];
        ks[t][d] = row[DMODEL + h * HDIM + d];
        vs[t][d] = row[2 * DMODEL + h * HDIM + d];
    }
    __syncthreads();
    for (int idx = tid; idx < LSEQQ * LSEQQ; idx += 256) {
        int i = idx / LSEQQ, j = idx % LSEQQ;
        float s = 0.0f;
        #pragma unroll
        for (int d = 0; d < HDIM; ++d) s += qs[i][d] * ks[j][d];
        sc[i][j] = s * 0.17677669529663687f;   // 1/sqrt(32)
    }
    __syncthreads();
    if (tid < LSEQQ) {
        int i = tid;
        float mx = -3.4e38f;
        for (int j = 0; j <= i; ++j) mx = fmaxf(mx, sc[i][j]);
        float sum = 0.0f;
        for (int j = 0; j < LSEQQ; ++j) {
            float p = (j <= i) ? expf(sc[i][j] - mx) : 0.0f;
            sc[i][j] = p; sum += p;
        }
        float inv = 1.0f / sum;
        for (int j = 0; j < LSEQQ; ++j) sc[i][j] *= inv;
    }
    __syncthreads();
    for (int idx = tid; idx < LSEQQ * HDIM; idx += 256) {
        int t = idx >> 5, d = idx & 31;
        float o = 0.0f;
        for (int j = 0; j <= t; ++j) o += sc[t][j] * vs[j][d];
        out[(size_t)(b * LSEQQ + t) * DMODEL + h * HDIM + d] = o;
    }
}

// cross-attention, block per (batch, head); q ld = 256, kv ld = 512 ([K|V])
__global__ __launch_bounds__(256)
void cross_attn_kernel(const float* __restrict__ q, const float* __restrict__ kv,
                       float* __restrict__ out)
{
    __shared__ float qs[LSEQQ][HDIM];
    __shared__ float ks[KCTX][HDIM];
    __shared__ float vs[KCTX][HDIM];
    __shared__ float sc[LSEQQ][KCTX];
    int b = blockIdx.x >> 3, h = blockIdx.x & 7;
    int tid = threadIdx.x;
    for (int idx = tid; idx < LSEQQ * HDIM; idx += 256) {
        int t = idx >> 5, d = idx & 31;
        qs[t][d] = q[(size_t)(b * LSEQQ + t) * DMODEL + h * HDIM + d];
    }
    for (int idx = tid; idx < KCTX * HDIM; idx += 256) {
        int t = idx >> 5, d = idx & 31;
        const float* row = kv + (size_t)(b * KCTX + t) * (2 * DMODEL);
        ks[t][d] = row[h * HDIM + d];
        vs[t][d] = row[DMODEL + h * HDIM + d];
    }
    __syncthreads();
    for (int idx = tid; idx < LSEQQ * KCTX; idx += 256) {
        int i = idx >> 7, j = idx & 127;
        float s = 0.0f;
        #pragma unroll
        for (int d = 0; d < HDIM; ++d) s += qs[i][d] * ks[j][d];
        sc[i][j] = s * 0.17677669529663687f;
    }
    __syncthreads();
    if (tid < LSEQQ) {
        float mx = -3.4e38f;
        for (int j = 0; j < KCTX; ++j) mx = fmaxf(mx, sc[tid][j]);
        float sum = 0.0f;
        for (int j = 0; j < KCTX; ++j) { float p = expf(sc[tid][j] - mx); sc[tid][j] = p; sum += p; }
        float inv = 1.0f / sum;
        for (int j = 0; j < KCTX; ++j) sc[tid][j] *= inv;
    }
    __syncthreads();
    for (int idx = tid; idx < LSEQQ * HDIM; idx += 256) {
        int t = idx >> 5, d = idx & 31;
        float o = 0.0f;
        for (int j = 0; j < KCTX; ++j) o += sc[t][j] * vs[j][d];
        out[(size_t)(b * LSEQQ + t) * DMODEL + h * HDIM + d] = o;
    }
}

__global__ void ctx_gather_kernel(const float* __restrict__ x, const int* __restrict__ ids,
                                  float* __restrict__ ctx)
{
    int i = blockIdx.x * 256 + threadIdx.x;
    if (i >= BATCH * KCTX * DMODEL) return;
    int d = i & 255, bk = i >> 8;
    ctx[i] = x[(size_t)ids[bk] * DMODEL + d];
}

__global__ void dec_init_kernel(const int* __restrict__ rid, const float* __restrict__ rel_emb,
                                const float* __restrict__ pos_emb, float* __restrict__ h)
{
    int i = blockIdx.x * 256 + threadIdx.x;
    if (i >= NTOK * DMODEL) return;
    int d = i & 255, bt = i >> 8, t = bt % LSEQQ;
    h[i] = rel_emb[(size_t)rid[bt] * DMODEL + d] + pos_emb[t * DMODEL + d];
}

// =====================================================================
// Host orchestration
// =====================================================================
extern "C" void kernel_launch(void* const* d_in, const int* in_sizes, int n_in,
                              void* d_out, int out_size, void* d_ws, size_t ws_size,
                              hipStream_t stream)
{
    const float* node_features = (const float*)d_in[0];
    const int*   edge_index    = (const int*)  d_in[1];
    const int*   relation_ids  = (const int*)  d_in[2];
    const int*   context_ids   = (const int*)  d_in[3];
    const float* in_proj_w     = (const float*)d_in[4];
    const float* in_proj_b     = (const float*)d_in[5];
    const float* gat_w         = (const float*)d_in[6];
    const float* gat_a         = (const float*)d_in[7];
    const float* enc_ln_g      = (const float*)d_in[8];
    const float* enc_ln_b      = (const float*)d_in[9];
    const float* rel_emb       = (const float*)d_in[10];
    const float* pos_emb       = (const float*)d_in[11];
    const float* sa_qkv_w      = (const float*)d_in[12];
    const float* sa_qkv_b      = (const float*)d_in[13];
    const float* sa_out_w      = (const float*)d_in[14];
    const float* sa_out_b      = (const float*)d_in[15];
    const float* ca_qkv_w      = (const float*)d_in[16];
    const float* ca_qkv_b      = (const float*)d_in[17];
    const float* ca_out_w      = (const float*)d_in[18];
    const float* ca_out_b      = (const float*)d_in[19];
    const float* n1_g = (const float*)d_in[20];
    const float* n1_b = (const float*)d_in[21];
    const float* n2_g = (const float*)d_in[22];
    const float* n2_b = (const float*)d_in[23];
    const float* n3_g = (const float*)d_in[24];
    const float* n3_b = (const float*)d_in[25];
    const float* ff_w1 = (const float*)d_in[26];
    const float* ff_b1 = (const float*)d_in[27];
    const float* ff_w2 = (const float*)d_in[28];
    const float* ff_b2 = (const float*)d_in[29];
    const float* head_w = (const float*)d_in[30];
    const float* head_b = (const float*)d_in[31];
    float* out = (float*)d_out;

    const int* src = edge_index;
    const int* tgt = edge_index + NEDGE;

    // ---- workspace carving ----
    char* ws = (char*)d_ws;
    size_t off = 0;
    auto alloc = [&](size_t bytes) -> void* {
        void* p = ws + off;
        off += bytes;
        off = (off + 255) & ~(size_t)255;
        return p;
    };
    float*    x      = (float*)   alloc((size_t)N_NODES * DMODEL * 4);
    float*    hfeat  = (float*)   alloc((size_t)N_NODES * DMODEL * 4);
    float*    gout   = (float*)   alloc((size_t)N_NODES * DMODEL * 4);
    float*    score  = (float*)   alloc((size_t)NEDGE * NHEAD * 4);
    unsigned* menc   = (unsigned*)alloc((size_t)N_NODES * NHEAD * 4);
    float*    sumex  = (float*)   alloc((size_t)N_NODES * NHEAD * 4);
    float*    denomb = (float*)   alloc((size_t)N_NODES * NHEAD * 4);
    unsigned* deg    = (unsigned*)alloc((size_t)N_NODES * 4);
    float*    s1     = (float*)   alloc((size_t)N_NODES * NHEAD * 4);
    float*    s2     = (float*)   alloc((size_t)N_NODES * NHEAD * 4);
    float*    ctx    = (float*)   alloc((size_t)BATCH * KCTX * DMODEL * 4);
    float*    dech   = (float*)   alloc((size_t)NTOK * DMODEL * 4);
    float*    hn     = (float*)   alloc((size_t)NTOK * DMODEL * 4);
    float*    qkvb   = (float*)   alloc((size_t)NTOK * 3 * DMODEL * 4);
    float*    attno  = (float*)   alloc((size_t)NTOK * DMODEL * 4);
    float*    qb     = (float*)   alloc((size_t)NTOK * DMODEL * 4);
    float*    kvb    = (float*)   alloc((size_t)BATCH * KCTX * 2 * DMODEL * 4);
    float*    ffb    = (float*)   alloc((size_t)NTOK * 4 * DMODEL * 4);

    auto gemm = [&](const float* A, const float* Wm, const float* bias, const float* resid,
                    float* C, int M, int Nn, int Kd, int epi) {
        dim3 grid((Nn + TILEN - 1) / TILEN, (M + TILEM - 1) / TILEM);
        gemm_wmma_kernel<<<grid, dim3(256), 0, stream>>>(A, Wm, bias, resid, C, M, Nn, Kd, epi);
    };

    // ---------------- GNN encoder ----------------
    gemm(node_features, in_proj_w, in_proj_b, nullptr, x, N_NODES, DMODEL, NODE_DIM, 0);

    for (int l = 0; l < N_GNN; ++l) {
        gemm(x, gat_w + (size_t)l * DMODEL * DMODEL, nullptr, nullptr,
             hfeat, N_NODES, DMODEL, DMODEL, 0);
        gat_score_kernel<<<N_NODES, 256, 0, stream>>>(hfeat, gat_a + l * 2 * HDIM, s1, s2);
        gat_init_kernel<<<(N_NODES * DMODEL + 255) / 256, 256, 0, stream>>>(menc, sumex, deg, gout);
        edge_score_kernel<<<(NEDGE * NHEAD + 255) / 256, 256, 0, stream>>>(
            src, tgt, s1, s2, score, menc, deg);
        gat_mclamp_kernel<<<(N_NODES * NHEAD + 255) / 256, 256, 0, stream>>>(menc);
        edge_exp_kernel<<<(NEDGE * NHEAD + 255) / 256, 256, 0, stream>>>(
            src, score, (const float*)menc, sumex);
        gat_denom_kernel<<<(N_NODES * NHEAD + 255) / 256, 256, 0, stream>>>(
            (const float*)menc, sumex, deg, denomb);
        gat_aggregate_kernel<<<NEDGE / 8, 256, 0, stream>>>(src, tgt, score, denomb, hfeat, gout);
        gat_post_kernel<<<N_NODES, 256, 0, stream>>>(x, gout, hfeat, deg, enc_ln_g, enc_ln_b);
    }

    ctx_gather_kernel<<<(BATCH * KCTX * DMODEL + 255) / 256, 256, 0, stream>>>(x, context_ids, ctx);

    // ---------------- decoder ----------------
    dec_init_kernel<<<(NTOK * DMODEL + 255) / 256, 256, 0, stream>>>(
        relation_ids, rel_emb, pos_emb, dech);

    for (int l = 0; l < N_DEC; ++l) {
        // self-attention block
        ln_rows_kernel<<<NTOK / 8, 256, 0, stream>>>(dech, hn, n1_g + l * DMODEL, n1_b + l * DMODEL, NTOK);
        gemm(hn, sa_qkv_w + (size_t)l * 3 * DMODEL * DMODEL, sa_qkv_b + l * 3 * DMODEL,
             nullptr, qkvb, NTOK, 3 * DMODEL, DMODEL, 0);
        self_attn_kernel<<<BATCH * NHEAD, 256, 0, stream>>>(qkvb, attno);
        gemm(attno, sa_out_w + (size_t)l * DMODEL * DMODEL, sa_out_b + l * DMODEL,
             dech, dech, NTOK, DMODEL, DMODEL, 0);

        // cross-attention block
        ln_rows_kernel<<<NTOK / 8, 256, 0, stream>>>(dech, hn, n2_g + l * DMODEL, n2_b + l * DMODEL, NTOK);
        gemm(hn, ca_qkv_w + (size_t)l * 3 * DMODEL * DMODEL, ca_qkv_b + l * 3 * DMODEL,
             nullptr, qb, NTOK, DMODEL, DMODEL, 0);
        gemm(ctx, ca_qkv_w + (size_t)l * 3 * DMODEL * DMODEL + (size_t)DMODEL * DMODEL,
             ca_qkv_b + l * 3 * DMODEL + DMODEL,
             nullptr, kvb, BATCH * KCTX, 2 * DMODEL, DMODEL, 0);
        cross_attn_kernel<<<BATCH * NHEAD, 256, 0, stream>>>(qb, kvb, attno);
        gemm(attno, ca_out_w + (size_t)l * DMODEL * DMODEL, ca_out_b + l * DMODEL,
             dech, dech, NTOK, DMODEL, DMODEL, 0);

        // FFN block
        ln_rows_kernel<<<NTOK / 8, 256, 0, stream>>>(dech, hn, n3_g + l * DMODEL, n3_b + l * DMODEL, NTOK);
        gemm(hn, ff_w1 + (size_t)l * 4 * DMODEL * DMODEL, ff_b1 + l * 4 * DMODEL,
             nullptr, ffb, NTOK, 4 * DMODEL, DMODEL, 1 /*gelu*/);
        gemm(ffb, ff_w2 + (size_t)l * 4 * DMODEL * DMODEL, ff_b2 + l * DMODEL,
             dech, dech, NTOK, DMODEL, 4 * DMODEL, 0);
    }

    // ---------------- output head ----------------
    gemm(dech, head_w, head_b, nullptr, out, NTOK, RREL, DMODEL, 0);
}
